// Transformer_7009386627625
// MI455X (gfx1250) — matrix operand order repaired and checked by reading, not compiled, not run on hardware
//
#include <hip/hip_runtime.h>
#include <hip/hip_bf16.h>
#include <hip/hip_fp16.h>

// ---------------- problem constants ----------------
constexpr int kB      = 2;
constexpr int kN      = 2048;
constexpr int kDim    = 1024;
constexpr int kHeads  = 16;
constexpr int kDH     = 64;
constexpr int kHid    = 4096;
constexpr int kDepth  = 6;
constexpr int kMTot   = kB * kN;          // 4096 tokens
constexpr float kScale = 0.125f;          // 64^-0.5
constexpr float kEps   = 1e-5f;

typedef __attribute__((ext_vector_type(16))) _Float16 v16h;
typedef __attribute__((ext_vector_type(8)))  float    v8f;

union HFrag { v16h v; unsigned int u[8]; _Float16 h[16]; };
union U64H4 { uint2 u2; _Float16 h[4]; };

__device__ __forceinline__ unsigned int ldu32(const _Float16* p) {
    return *reinterpret_cast<const unsigned int*>(p);
}
__device__ __forceinline__ uint2 ldu64(const _Float16* p) {
    return *reinterpret_cast<const uint2*>(p);
}
__device__ __forceinline__ void stu64(_Float16* p, uint2 v) {
    *reinterpret_cast<uint2*>(p) = v;
}

// A-matrix 16x32 f16 fragment per CDNA5 ISA layout:
// lanes 0-15: V0..3 = K 0..7 pairs, V4..7 = K 16..23 pairs (M = lane)
// lanes 16-31: V0..3 = K 8..15 pairs, V4..7 = K 24..31 pairs
__device__ __forceinline__ v16h load_frag_a(const _Float16* base, int stride,
                                            int m, int k0, int hs) {
    HFrag f;
    const _Float16* row = base + m * stride + k0;
#pragma unroll
    for (int vi = 0; vi < 4; ++vi) {
        f.u[vi]     = ldu32(row + hs * 8 + 2 * vi);
        f.u[4 + vi] = ldu32(row + 16 + hs * 8 + 2 * vi);
    }
    return f.v;
}

// B-matrix 32x16 f16 fragment: lane&15 = N, same K layout as A:
// lanes 0-15: V0..3 = K 0..7, V4..7 = K 16..23; lanes 16-31: K 8..15, 24..31.  base indexed [n][k].
__device__ __forceinline__ v16h load_frag_b(const _Float16* base, int stride,
                                            int n, int k0, int hs) {
    HFrag f;
    const _Float16* row = base + n * stride + k0;
#pragma unroll
    for (int vi = 0; vi < 4; ++vi) {
        f.u[vi]     = ldu32(row + hs * 8 + 2 * vi);
        f.u[4 + vi] = ldu32(row + 16 + hs * 8 + 2 * vi);
    }
    return f.v;
}

// ---------------- f32 -> f16 convert (4 elems/thread) ----------------
__global__ void __launch_bounds__(256)
cvt_f32_f16(const float* __restrict__ in, _Float16* __restrict__ out, int n4) {
    int i = blockIdx.x * 256 + threadIdx.x;
    if (i < n4) {
        float4 v = reinterpret_cast<const float4*>(in)[i];
        U64H4 o;
        o.h[0] = (_Float16)v.x; o.h[1] = (_Float16)v.y;
        o.h[2] = (_Float16)v.z; o.h[3] = (_Float16)v.w;
        reinterpret_cast<uint2*>(out)[i] = o.u2;
    }
}

// ---------------- LayerNorm (one block per token) ----------------
__global__ void __launch_bounds__(256)
ln_kernel(const float* __restrict__ x, const float* __restrict__ g,
          const float* __restrict__ b, _Float16* __restrict__ out) {
    __shared__ float s1[8], s2[8];
    const int row = blockIdx.x;
    const int t = threadIdx.x;
    const float* xr = x + (size_t)row * kDim;
    float4 v = reinterpret_cast<const float4*>(xr)[t];
    float sum = v.x + v.y + v.z + v.w;
    float sq  = v.x * v.x + v.y * v.y + v.z * v.z + v.w * v.w;
#pragma unroll
    for (int off = 16; off; off >>= 1) {
        sum += __shfl_xor(sum, off, 32);
        sq  += __shfl_xor(sq,  off, 32);
    }
    if ((t & 31) == 0) { s1[t >> 5] = sum; s2[t >> 5] = sq; }
    __syncthreads();
    float tot = 0.f, tot2 = 0.f;
#pragma unroll
    for (int i = 0; i < 8; ++i) { tot += s1[i]; tot2 += s2[i]; }
    const float mean = tot * (1.0f / kDim);
    const float var  = tot2 * (1.0f / kDim) - mean * mean;
    const float inv  = rsqrtf(var + kEps);
    const float4 gv = reinterpret_cast<const float4*>(g)[t];
    const float4 bv = reinterpret_cast<const float4*>(b)[t];
    U64H4 o;
    o.h[0] = (_Float16)((v.x - mean) * inv * gv.x + bv.x);
    o.h[1] = (_Float16)((v.y - mean) * inv * gv.y + bv.y);
    o.h[2] = (_Float16)((v.z - mean) * inv * gv.z + bv.z);
    o.h[3] = (_Float16)((v.w - mean) * inv * gv.w + bv.w);
    reinterpret_cast<uint2*>(out + (size_t)row * kDim)[t] = o.u2;
}

// ---------------- WMMA GEMM: C[M,N] = A[M,K](f16) @ B[K,N](f16) + epilogue ---
// Block tile 128x128xBK32, 8 waves in 4(M)x2(N) grid, wave tile 32x64,
// double-buffered LDS (one barrier per k-step), 8 WMMA per wave per k-step.
// Staging uses b64 global loads; b64 LDS stores for A, 16-bit transposing stores for B.
// MODE 0: +bias, split into q(scaled)/k/v f16 head-major buffers
// MODE 1: +bias +fp32 residual -> fp32 out
// MODE 2: +bias, exact GELU -> f16 out
template <int MODE>
__global__ void __launch_bounds__(256)
gemm_wmma(const _Float16* __restrict__ A, const _Float16* __restrict__ Bw,
          const float* __restrict__ bias, const float* __restrict__ res,
          float* __restrict__ outF, _Float16* __restrict__ outH,
          _Float16* __restrict__ qh, _Float16* __restrict__ kh,
          _Float16* __restrict__ vh, int M, int Nn, int K) {
    __shared__ _Float16 sA[2][128 * 36];   // [m][k] padded (72B rows)
    __shared__ _Float16 sB[2][128 * 36];   // [n][k] padded (transposed tile)
    const int t = threadIdx.x;
    const int lane = t & 31, w = t >> 5;
    const int hs = lane >> 4, l15 = lane & 15;
    const int bm = blockIdx.y * 128, bn = blockIdx.x * 128;
    const int wm = (w >> 1) * 32, wn = (w & 1) * 64;

    v8f acc[2][4] = {};
    const int ksteps = K >> 5;

    // A tile: 128 rows x 32 halves = 1024 u64 chunks -> 4 per thread
    // chunk j: row = j>>3, c4 = j&7 (4-half groups)
    // B tile: 32 k-rows x 128 halves = 1024 u64 chunks -> 4 per thread
    // chunk j: kk = j>>5, n4 = j&31 (4 n-values at one k)

    // ---- prologue: stage tile 0 into buffer 0 ----
    {
#pragma unroll
        for (int i = 0; i < 4; ++i) {
            int j = t + i * 256;
            int r = j >> 3, c4 = j & 7;
            uint2 u = ldu64(A + (size_t)(bm + r) * K + c4 * 4);
            stu64(&sA[0][r * 36 + c4 * 4], u);
        }
#pragma unroll
        for (int i = 0; i < 4; ++i) {
            int j = t + i * 256;
            int kk = j >> 5, n4 = j & 31;
            U64H4 cv; cv.u2 = ldu64(Bw + (size_t)kk * Nn + bn + n4 * 4);
#pragma unroll
            for (int e = 0; e < 4; ++e)
                sB[0][(n4 * 4 + e) * 36 + kk] = cv.h[e];
        }
    }
    __syncthreads();

    for (int kt = 0; kt < ksteps; ++kt) {
        const int cur = kt & 1, nxt = cur ^ 1;
        uint2 aReg[4], bReg[4];
        const bool more = (kt + 1 < ksteps);
        if (more) {
#pragma unroll
            for (int i = 0; i < 4; ++i) {
                int j = t + i * 256;
                aReg[i] = ldu64(A + (size_t)(bm + (j >> 3)) * K +
                                (kt + 1) * 32 + (j & 7) * 4);
            }
#pragma unroll
            for (int i = 0; i < 4; ++i) {
                int j = t + i * 256;
                bReg[i] = ldu64(Bw + (size_t)((kt + 1) * 32 + (j >> 5)) * Nn +
                                bn + (j & 31) * 4);
            }
            if (kt + 2 < ksteps)
                __builtin_prefetch(A + (size_t)(bm + (t >> 1)) * K + (kt + 2) * 32, 0, 1);
        }

        // ---- compute on current buffer: 8 WMMA per wave ----
        v16h af[2];
#pragma unroll
        for (int mt = 0; mt < 2; ++mt)
            af[mt] = load_frag_a(sA[cur], 36, wm + mt * 16 + l15, 0, hs);
#pragma unroll
        for (int nt = 0; nt < 4; ++nt) {
            v16h bf = load_frag_b(sB[cur], 36, wn + nt * 16 + l15, 0, hs);
#pragma unroll
            for (int mt = 0; mt < 2; ++mt)
                acc[mt][nt] = __builtin_amdgcn_wmma_f32_16x16x32_f16(
                    false, af[mt], false, bf, (short)0, acc[mt][nt],
                    false, false);
        }

        if (more) {
#pragma unroll
            for (int i = 0; i < 4; ++i) {
                int j = t + i * 256;
                stu64(&sA[nxt][(j >> 3) * 36 + (j & 7) * 4], aReg[i]);
            }
#pragma unroll
            for (int i = 0; i < 4; ++i) {
                int j = t + i * 256;
                int kk = j >> 5, n4 = j & 31;
                U64H4 cv; cv.u2 = bReg[i];
#pragma unroll
                for (int e = 0; e < 4; ++e)
                    sB[nxt][(n4 * 4 + e) * 36 + kk] = cv.h[e];
            }
            __syncthreads();
        }
    }

    // ---- epilogue ----
#pragma unroll
    for (int mt = 0; mt < 2; ++mt) {
#pragma unroll
        for (int nt = 0; nt < 4; ++nt) {
            const int gn = bn + wn + nt * 16 + l15;
#pragma unroll
            for (int r = 0; r < 8; ++r) {
                const int gm = bm + wm + mt * 16 + hs * 8 + r;
                float vv = acc[mt][nt][r] + bias[gn];
                if (MODE == 0) {
                    const int part = gn >> 10;
                    const int j = gn & 1023;
                    const int head = j >> 6, dd = j & 63;
                    const int bb = gm >> 11, nrow = gm & (kN - 1);
                    const size_t oi =
                        (((size_t)(bb * kHeads + head)) * kN + nrow) * kDH + dd;
                    if (part == 0)      qh[oi] = (_Float16)(vv * kScale);
                    else if (part == 1) kh[oi] = (_Float16)vv;
                    else                vh[oi] = (_Float16)vv;
                } else if (MODE == 1) {
                    const size_t oi = (size_t)gm * Nn + gn;
                    outF[oi] = vv + res[oi];
                } else {
                    const float ge = 0.5f * vv * (1.0f + erff(vv * 0.70710678118654752f));
                    outH[(size_t)gm * Nn + gn] = (_Float16)ge;
                }
            }
        }
    }
}

// ---------------- Flash attention: block = (b*h, 128 q rows), 8 waves -------
// 64-key chunks: 8 WMMA for S (K=64 chained), 8 WMMA for O += P@V per chunk.
__global__ void __launch_bounds__(256)
attn_kernel(const _Float16* __restrict__ q, const _Float16* __restrict__ k,
            const _Float16* __restrict__ v, _Float16* __restrict__ o) {
    __shared__ _Float16 sK[64 * 68];      // [key][dh] padded (136B rows)
    __shared__ _Float16 sVt[64 * 68];     // [dh][key] padded (transposed)
    __shared__ _Float16 sP[8 * 16 * 68];  // per-wave P tile [row][key]
    const int t = threadIdx.x, lane = t & 31, w = t >> 5;
    const int hs = lane >> 4, l15 = lane & 15;
    const int bh = blockIdx.x;                       // b*Heads + h
    const int qbase = blockIdx.y * 128 + w * 16;     // first q row of this wave

    // Q fragments (dh = 64 -> two K=32 blocks), loaded straight from global
    v16h qa[2];
    {
        const _Float16* qr = q + ((size_t)bh * kN + qbase + l15) * kDH;
#pragma unroll
        for (int kb = 0; kb < 2; ++kb) {
            HFrag f;
#pragma unroll
            for (int vi = 0; vi < 4; ++vi) {
                f.u[vi]     = ldu32(qr + kb * 32 + hs * 8 + 2 * vi);
                f.u[4 + vi] = ldu32(qr + kb * 32 + 16 + hs * 8 + 2 * vi);
            }
            qa[kb] = f.v;
        }
    }

    v8f oacc[4] = {};
    float mrow[8], lrow[8];
#pragma unroll
    for (int r = 0; r < 8; ++r) { mrow[r] = -3.0e38f; lrow[r] = 0.f; }

    const _Float16* kbase = k + (size_t)bh * kN * kDH;
    const _Float16* vbase = v + (size_t)bh * kN * kDH;

    for (int ch = 0; ch < kN / 64; ++ch) {
        __syncthreads();
        // stage 64 keys x 64 dh of K and V (V transposed): 1024 u64 each
        // chunk j: key = j>>4, d4 = j&15 (4-half groups)
#pragma unroll
        for (int i = 0; i < 4; ++i) {
            int j = t + i * 256;
            int key = j >> 4, d4 = j & 15;
            uint2 uK = ldu64(kbase + (size_t)(ch * 64 + key) * kDH + d4 * 4);
            stu64(&sK[key * 68 + d4 * 4], uK);
            U64H4 cv;
            cv.u2 = ldu64(vbase + (size_t)(ch * 64 + key) * kDH + d4 * 4);
#pragma unroll
            for (int e = 0; e < 4; ++e)
                sVt[(d4 * 4 + e) * 68 + key] = cv.h[e];
        }
        __syncthreads();

        // S = Q @ K^T  (16 rows x 64 keys per wave)
        v8f s[4] = {};
#pragma unroll
        for (int nt = 0; nt < 4; ++nt) {
#pragma unroll
            for (int kb = 0; kb < 2; ++kb) {
                v16h bf = load_frag_b(sK, 68, nt * 16 + l15, kb * 32, hs);
                s[nt] = __builtin_amdgcn_wmma_f32_16x16x32_f16(
                    false, qa[kb], false, bf, (short)0, s[nt], false, false);
            }
        }

        // online softmax (per-row stats, 16-lane reductions within halves)
        float alpha[8];
#pragma unroll
        for (int r = 0; r < 8; ++r) {
            float tmax = fmaxf(fmaxf(s[0][r], s[1][r]), fmaxf(s[2][r], s[3][r]));
#pragma unroll
            for (int off = 1; off < 16; off <<= 1)
                tmax = fmaxf(tmax, __shfl_xor(tmax, off, 32));
            const float nm = fmaxf(mrow[r], tmax);
            alpha[r] = __expf(mrow[r] - nm);
            mrow[r] = nm;
            float rs = 0.f;
#pragma unroll
            for (int nt = 0; nt < 4; ++nt) {
                const float pv = __expf(s[nt][r] - nm);
                s[nt][r] = pv;
                rs += pv;
            }
#pragma unroll
            for (int off = 1; off < 16; off <<= 1)
                rs += __shfl_xor(rs, off, 32);
            lrow[r] = lrow[r] * alpha[r] + rs;
        }

        // write P (C-layout) to wave-private LDS region, re-read in A-layout.
        // Same-wave LDS ops execute in order -> no workgroup barrier needed.
        _Float16* pW = sP + w * 16 * 68;
#pragma unroll
        for (int nt = 0; nt < 4; ++nt)
#pragma unroll
            for (int r = 0; r < 8; ++r)
                pW[(hs * 8 + r) * 68 + nt * 16 + l15] = (_Float16)s[nt][r];

        // rescale O accumulators
#pragma unroll
        for (int j = 0; j < 4; ++j)
#pragma unroll
            for (int r = 0; r < 8; ++r) oacc[j][r] = oacc[j][r] * alpha[r];

        v16h pf[2];
        {
            const _Float16* pr = pW + l15 * 68;
#pragma unroll
            for (int kb = 0; kb < 2; ++kb) {
                HFrag f;
#pragma unroll
                for (int vi = 0; vi < 4; ++vi) {
                    f.u[vi]     = ldu32(pr + kb * 32 + hs * 8 + 2 * vi);
                    f.u[4 + vi] = ldu32(pr + kb * 32 + 16 + hs * 8 + 2 * vi);
                }
                pf[kb] = f.v;
            }
        }
        // O += P @ V  (K = 64 keys, chained)
#pragma unroll
        for (int j = 0; j < 4; ++j) {
#pragma unroll
            for (int kb = 0; kb < 2; ++kb) {
                v16h vf = load_frag_b(sVt, 68, j * 16 + l15, kb * 32, hs);
                oacc[j] = __builtin_amdgcn_wmma_f32_16x16x32_f16(
                    false, pf[kb], false, vf, (short)0, oacc[j], false, false);
            }
        }
    }

    // finalize: O /= l, write f16 token-major [token][h*64 + dh]
    const int bb = bh / kHeads, hh = bh % kHeads;
#pragma unroll
    for (int j = 0; j < 4; ++j)
#pragma unroll
        for (int r = 0; r < 8; ++r) {
            const int rowTok = bb * kN + qbase + hs * 8 + r;
            const int col = hh * kDH + j * 16 + l15;
            o[(size_t)rowTok * kDim + col] = (_Float16)(oacc[j][r] / lrow[r]);
        }
}

// ---------------- host launcher ----------------
extern "C" void kernel_launch(void* const* d_in, const int* in_sizes, int n_in,
                              void* d_out, int out_size, void* d_ws, size_t ws_size,
                              hipStream_t stream) {
    const float* x    = (const float*)d_in[0];
    const float* Wqkv = (const float*)d_in[1];
    const float* bqkv = (const float*)d_in[2];
    const float* Wo   = (const float*)d_in[3];
    const float* bo   = (const float*)d_in[4];
    const float* W1   = (const float*)d_in[5];
    const float* bm1  = (const float*)d_in[6];
    const float* W2   = (const float*)d_in[7];
    const float* bm2  = (const float*)d_in[8];
    const float* g1   = (const float*)d_in[9];
    const float* be1  = (const float*)d_in[10];
    const float* g2   = (const float*)d_in[11];
    const float* be2  = (const float*)d_in[12];
    float* xbuf = (float*)d_out;

    char* p = (char*)d_ws;
    auto carve = [&](size_t halves) -> _Float16* {
        _Float16* r = (_Float16*)p;
        p += halves * sizeof(_Float16);
        return r;
    };
    _Float16* wqkv16 = carve((size_t)kDim * 3 * kDim);
    _Float16* wo16   = carve((size_t)kDim * kDim);
    _Float16* w116   = carve((size_t)kDim * kHid);
    _Float16* w216   = carve((size_t)kHid * kDim);
    _Float16* h16    = carve((size_t)kMTot * kDim);
    _Float16* q16    = carve((size_t)kMTot * kDim);
    _Float16* k16    = carve((size_t)kMTot * kDim);
    _Float16* vv16   = carve((size_t)kMTot * kDim);
    _Float16* o16    = carve((size_t)kMTot * kDim);
    _Float16* m16    = carve((size_t)kMTot * kHid);

    hipMemcpyAsync(xbuf, x, (size_t)kMTot * kDim * sizeof(float),
                   hipMemcpyDeviceToDevice, stream);

    const int nq = kDim * 3 * kDim, no = kDim * kDim;
    const int n1 = kDim * kHid, n2 = kHid * kDim;

    for (int d = 0; d < kDepth; ++d) {
        cvt_f32_f16<<<(nq / 4 + 255) / 256, 256, 0, stream>>>(Wqkv + (size_t)d * nq, wqkv16, nq / 4);
        cvt_f32_f16<<<(no / 4 + 255) / 256, 256, 0, stream>>>(Wo   + (size_t)d * no, wo16,   no / 4);
        cvt_f32_f16<<<(n1 / 4 + 255) / 256, 256, 0, stream>>>(W1   + (size_t)d * n1, w116,   n1 / 4);
        cvt_f32_f16<<<(n2 / 4 + 255) / 256, 256, 0, stream>>>(W2   + (size_t)d * n2, w216,   n2 / 4);

        // h = LN1(x)
        ln_kernel<<<kMTot, 256, 0, stream>>>(xbuf, g1 + (size_t)d * kDim,
                                             be1 + (size_t)d * kDim, h16);
        // qkv = h @ Wqkv + bqkv  -> q(scaled)/k/v f16, head-major
        gemm_wmma<0><<<dim3(3 * kDim / 128, kMTot / 128), 256, 0, stream>>>(
            h16, wqkv16, bqkv + (size_t)d * 3 * kDim, nullptr, nullptr, nullptr,
            q16, k16, vv16, kMTot, 3 * kDim, kDim);
        // attention
        attn_kernel<<<dim3(kB * kHeads, kN / 128), 256, 0, stream>>>(q16, k16, vv16, o16);
        // x = O @ Wo + bo + x
        gemm_wmma<1><<<dim3(kDim / 128, kMTot / 128), 256, 0, stream>>>(
            o16, wo16, bo + (size_t)d * kDim, xbuf, xbuf, nullptr,
            nullptr, nullptr, nullptr, kMTot, kDim, kDim);
        // h = LN2(x)
        ln_kernel<<<kMTot, 256, 0, stream>>>(xbuf, g2 + (size_t)d * kDim,
                                             be2 + (size_t)d * kDim, h16);
        // m = GELU(h @ W1 + b1) (f16)
        gemm_wmma<2><<<dim3(kHid / 128, kMTot / 128), 256, 0, stream>>>(
            h16, w116, bm1 + (size_t)d * kHid, nullptr, nullptr, m16,
            nullptr, nullptr, nullptr, kMTot, kHid, kDim);
        // x = m @ W2 + b2 + x
        gemm_wmma<1><<<dim3(kDim / 128, kMTot / 128), 256, 0, stream>>>(
            m16, w216, bm2 + (size_t)d * kDim, xbuf, xbuf, nullptr,
            nullptr, nullptr, nullptr, kMTot, kDim, kHid);
    }
}
